// BasicBlock9_4947802325432
// MI455X (gfx1250) — matrix operand order repaired and checked by reading, not compiled, not run on hardware
//
#include <hip/hip_runtime.h>

typedef __attribute__((ext_vector_type(16))) _Float16 v16h;
typedef __attribute__((ext_vector_type(8)))  float    v8f;

#define NB 16
#define NC 64
#define NH 64
#define NW 64
#define NHW  (NH*NW)
#define NCHW (NC*NHW)
#define NOFF 18          // 2*3*3 offset channels

// ---------------------------------------------------------------------------
// Fold batchnorm params into per-channel scale/bias:  y = conv*s + o
// sb layout: [s1|o1|s3|o3|s2|o2], 64 floats each.
// ---------------------------------------------------------------------------
__global__ __launch_bounds__(64) void bn_fold_kernel(
    const float* g1, const float* b1, const float* m1, const float* v1,
    const float* g3, const float* b3, const float* m3, const float* v3,
    const float* g2, const float* b2, const float* m2, const float* v2,
    float* sb)
{
    int i = threadIdx.x;
    if (i < NC) {
        float s;
        s = g1[i]*rsqrtf(v1[i]+1e-5f); sb[0*NC+i]=s; sb[1*NC+i]=b1[i]-m1[i]*s;
        s = g3[i]*rsqrtf(v3[i]+1e-5f); sb[2*NC+i]=s; sb[3*NC+i]=b3[i]-m3[i]*s;
        s = g2[i]*rsqrtf(v2[i]+1e-5f); sb[4*NC+i]=s; sb[5*NC+i]=b2[i]-m2[i]*s;
    }
}

// ---------------------------------------------------------------------------
// Pre-pack a [64 x Ktot] f32 weight matrix into wave32 WMMA A-fragments, f16.
// Fragment f = ((mtile*KT + kt)*32 + lane) holds the 16 halves lane `lane`
// feeds to v_wmma for M-tile `mtile`, K-chunk `kt` (ISA 7.12.2 16-bit A
// layout: lanes 0-15 K={0..7,16..23}, lanes 16-31 K={8..15,24..31}).
// ---------------------------------------------------------------------------
__global__ __launch_bounds__(256) void wpack_kernel(
    const float* __restrict__ w, _Float16* __restrict__ out, int Ktot)
{
    int tid = blockIdx.x*256 + threadIdx.x;
    int KT = Ktot >> 5;
    int nfrag = 4*KT*32;
    if (tid >= nfrag) return;
    int lane = tid & 31;
    int kt   = (tid>>5) % KT;
    int mt   = (tid>>5) / KT;
    int m    = mt*16 + (lane & 15);
    int khalf = (lane >> 4) * 8;
    int kc    = kt*32;
    const float* wr = w + (size_t)m*Ktot;
    v16h a;
    #pragma unroll
    for (int r = 0; r < 4; ++r) {
        int k0 = kc + khalf + 2*r;
        a[2*r  ] = (_Float16)wr[k0];
        a[2*r+1] = (_Float16)wr[k0+1];
        int k1 = kc + 16 + khalf + 2*r;
        a[2*r+8] = (_Float16)wr[k1];
        a[2*r+9] = (_Float16)wr[k1+1];
    }
    ((v16h*)out)[tid] = a;
}

// ---------------------------------------------------------------------------
// Offset conv: 3x3, Cin=64 -> 18 channels, pad 1, + bias.  (f32 VALU; small,
// and its output feeds floor() so keep it full precision.)
// ---------------------------------------------------------------------------
__global__ __launch_bounds__(128) void offset_conv_kernel(
    const float* __restrict__ x, const float* __restrict__ w_off,
    const float* __restrict__ b_off, float* __restrict__ off1)
{
    __shared__ float lx[NC*3*18];
    int bid = blockIdx.x;
    int wt = bid & 3, h = (bid>>2)&63, b = bid>>8;
    int w0 = wt*16;
    int tid = threadIdx.x;
    for (int e = tid; e < NC*3*18; e += 128) {
        int ci = e/54, r = e%54, ky = r/18, col = r%18;
        int y = h + ky - 1, xc = w0 + col - 1;
        float v = 0.f;
        if (y>=0 && y<NH && xc>=0 && xc<NW)
            v = x[b*NCHW + ci*NHW + y*NW + xc];
        lx[e] = v;
    }
    __syncthreads();
    for (int o = tid; o < NOFF*16; o += 128) {
        int co = o>>4, n = o&15;
        float acc = b_off[co];
        const float* wp = w_off + co*NC*9;
        for (int ci = 0; ci < NC; ++ci) {
            const float* lp = lx + ci*54;
            #pragma unroll
            for (int ky = 0; ky < 3; ++ky)
                #pragma unroll
                for (int kx = 0; kx < 3; ++kx)
                    acc += wp[ci*9 + ky*3 + kx] * lp[ky*18 + n + kx];
        }
        off1[b*NOFF*NHW + co*NHW + h*NW + w0 + n] = acc;
    }
}

// Shared WMMA mainloop: A from packed global fragments, B from paired LDS.
// patchw word layout: word[(k>>1)*16 + n] = halves { B(k,n), B(k+1,n) }.
__device__ __forceinline__ v8f wmma_mainloop(
    const v16h* __restrict__ apack, const unsigned int* patchw,
    int KT, int lane, int wv)
{
    int nn = lane & 15;
    int kb = (lane >> 4) * 8;
    const v16h* ap = apack + (size_t)wv*KT*32 + lane;
    v8f c = {};
    for (int kt = 0; kt < KT; ++kt) {
        v16h a = ap[(size_t)kt*32];
        union { v16h h; unsigned int u[8]; } bfu;
        int basew = ((kt*32 + kb) >> 1)*16 + nn;
        #pragma unroll
        for (int r = 0; r < 8; ++r)
            bfu.u[r] = patchw[basew + (r + (r >= 4 ? 4 : 0))*16];
        c = __builtin_amdgcn_wmma_f32_16x16x32_f16(false, a, false, bfu.h,
                                                   (short)0, c, false, false);
    }
    return c;
}

// ---------------------------------------------------------------------------
// Deformable conv 3x3 (K=576 implicit GEMM) + BN1 + relu -> out.
// ---------------------------------------------------------------------------
__global__ __launch_bounds__(128) void deform_wmma_kernel(
    const float* __restrict__ x, const float* __restrict__ off1,
    const v16h* __restrict__ w1p, const float* __restrict__ sb,
    float* __restrict__ out)
{
    __shared__ int      p_y0[144], p_x0[144];
    __shared__ float    p_w00[144], p_w01[144], p_w10[144], p_w11[144];
    __shared__ unsigned int valsw[(576/2)*16];   // paired f16 [k/2][n][2]

    int bid = blockIdx.x;
    int wt = bid & 3, h = (bid>>2)&63, b = bid>>8;
    int w0 = wt*16;
    int tid = threadIdx.x;

    const float* offb = off1 + b*NOFF*NHW + h*NW + w0;
    for (int t = tid; t < 144; t += 128) {
        int kk = t>>4, n = t&15;
        int ky = kk/3, kx = kk%3;
        float dy = offb[(2*kk  )*NHW + n];
        float dx = offb[(2*kk+1)*NHW + n];
        float ys = (float)(h + ky - 1) + dy;
        float xs = (float)(w0 + n + kx - 1) + dx;
        float y0f = floorf(ys), x0f = floorf(xs);
        float wy1 = ys - y0f,  wx1 = xs - x0f;
        float wy0 = 1.f - wy1, wx0 = 1.f - wx1;
        p_y0[t] = (int)y0f; p_x0[t] = (int)x0f;
        p_w00[t] = wy0*wx0; p_w01[t] = wy0*wx1;
        p_w10[t] = wy1*wx0; p_w11[t] = wy1*wx1;
    }
    __syncthreads();

    _Float16* vh = (_Float16*)valsw;
    for (int e = tid; e < NC*144; e += 128) {
        int ci = e/144, t = e - ci*144;
        const float* xp = x + b*NCHW + ci*NHW;
        int y0 = p_y0[t], x0i = p_x0[t];
        float acc = 0.f;
        { int yi=y0,   xi=x0i;   if (yi>=0 && yi<NH && xi>=0 && xi<NW) acc += p_w00[t]*xp[yi*NW+xi]; }
        { int yi=y0,   xi=x0i+1; if (yi>=0 && yi<NH && xi>=0 && xi<NW) acc += p_w01[t]*xp[yi*NW+xi]; }
        { int yi=y0+1, xi=x0i;   if (yi>=0 && yi<NH && xi>=0 && xi<NW) acc += p_w10[t]*xp[yi*NW+xi]; }
        { int yi=y0+1, xi=x0i+1; if (yi>=0 && yi<NH && xi>=0 && xi<NW) acc += p_w11[t]*xp[yi*NW+xi]; }
        int kk = t>>4, n = t&15;
        int k = ci*9 + kk;
        vh[(k>>1)*32 + 2*n + (k&1)] = (_Float16)acc;
    }
    __syncthreads();

    int lane = tid & 31, wv = tid >> 5;
    v8f c = wmma_mainloop(w1p, valsw, 576/32, lane, wv);

    int nn = lane & 15;
    float* ob = out + b*NCHW + h*NW + w0;
    #pragma unroll
    for (int r = 0; r < 8; ++r) {
        int co = wv*16 + r + 8*(lane>>4);
        float v = c[r]*sb[co] + sb[NC+co];
        ob[co*NHW + nn] = v > 0.f ? v : 0.f;
    }
}

// ---------------------------------------------------------------------------
// Generic conv KSxKS (stride 1, pad (KS-1)/2) as WMMA implicit GEMM + BN.
// MODE 1: out += relu(bn(conv(in)))           (5x5 branch, accumulate)
// MODE 2: out  = relu(bn(conv(in)) + res)     (final 3x3 + residual)
// ---------------------------------------------------------------------------
template<int KS, int MODE>
__global__ __launch_bounds__(128) void conv_wmma_kernel(
    const float* __restrict__ in, const v16h* __restrict__ wpk,
    const float* __restrict__ sb, const float* __restrict__ res,
    float* __restrict__ out)
{
    constexpr int PAD  = (KS-1)/2;
    constexpr int COLS = 16 + 2*PAD;
    constexpr int KK   = KS*KS;
    constexpr int KTOT = NC*KK;
    __shared__ _Float16     lx[NC*KS*COLS];
    __shared__ unsigned int patchw[(KTOT/2)*16];   // paired f16 [k/2][n][2]

    int bid = blockIdx.x;
    int wt = bid & 3, h = (bid>>2)&63, b = bid>>8;
    int w0 = wt*16;
    int tid = threadIdx.x;

    for (int e = tid; e < NC*KS*COLS; e += 128) {
        int ci = e/(KS*COLS), r = e - ci*(KS*COLS);
        int ky = r/COLS, col = r - ky*COLS;
        int y = h + ky - PAD, xc = w0 + col - PAD;
        float v = 0.f;
        if (y>=0 && y<NH && xc>=0 && xc<NW)
            v = in[b*NCHW + ci*NHW + y*NW + xc];
        lx[e] = (_Float16)v;
    }
    __syncthreads();

    // im2col into paired LDS layout (hoists all k->(ci,ky,kx) math here)
    _Float16* ph = (_Float16*)patchw;
    for (int e = tid; e < KTOT*16; e += 128) {
        int k = e>>4, n = e&15;
        int ci = k/KK;  int rem = k - ci*KK;
        int ky = rem/KS; int kx = rem - ky*KS;
        ph[(k>>1)*32 + 2*n + (k&1)] = lx[ci*(KS*COLS) + ky*COLS + n + kx];
    }
    __syncthreads();

    int lane = tid & 31, wv = tid >> 5;
    v8f c = wmma_mainloop(wpk, patchw, KTOT/32, lane, wv);

    int nn = lane & 15;
    float* ob = out + b*NCHW + h*NW + w0;
    #pragma unroll
    for (int r = 0; r < 8; ++r) {
        int co  = wv*16 + r + 8*(lane>>4);
        int idx = co*NHW + nn;
        float v = c[r]*sb[co] + sb[NC+co];
        if constexpr (MODE == 1) {
            v = v > 0.f ? v : 0.f;
            ob[idx] = ob[idx] + v;
        } else {
            v += res[b*NCHW + h*NW + w0 + idx];
            ob[idx] = v > 0.f ? v : 0.f;
        }
    }
}

// ---------------------------------------------------------------------------
// Launch: bn-fold + weight pre-pack -> offset conv -> deform WMMA ->
//         5x5 WMMA (accum) -> final 3x3 WMMA (+x residual, relu) -> d_out.
// Workspace (floats): sb 384 | off1 1179648 | outa 4194304 | packed f16
// weights (w1p 36864h, w3p 102400h, w2p 36864h)  ~= 21.9 MB total.
// ---------------------------------------------------------------------------
extern "C" void kernel_launch(void* const* d_in, const int* in_sizes, int n_in,
                              void* d_out, int out_size, void* d_ws, size_t ws_size,
                              hipStream_t stream) {
    const float* x     = (const float*)d_in[0];
    const float* w_off = (const float*)d_in[1];
    const float* b_off = (const float*)d_in[2];
    const float* w1    = (const float*)d_in[3];
    const float* g1    = (const float*)d_in[4];
    const float* b1    = (const float*)d_in[5];
    const float* m1    = (const float*)d_in[6];
    const float* v1    = (const float*)d_in[7];
    const float* w3    = (const float*)d_in[8];
    const float* g3    = (const float*)d_in[9];
    const float* b3    = (const float*)d_in[10];
    const float* m3    = (const float*)d_in[11];
    const float* v3    = (const float*)d_in[12];
    const float* w2    = (const float*)d_in[13];
    const float* g2    = (const float*)d_in[14];
    const float* b2    = (const float*)d_in[15];
    const float* m2    = (const float*)d_in[16];
    const float* v2    = (const float*)d_in[17];
    float* out = (float*)d_out;

    float* sb   = (float*)d_ws;
    float* off1 = sb + 6*NC;                       // 16*18*4096 floats
    float* outa = off1 + (size_t)NB*NOFF*NHW;      // 16*64*4096 floats
    _Float16* w1p = (_Float16*)(outa + (size_t)NB*NC*NHW);  // 32B-aligned
    _Float16* w3p = w1p + (size_t)NC*576;
    _Float16* w2p = w3p + (size_t)NC*1600;

    const int nblk = NB * NH * (NW/16);   // 4096

    bn_fold_kernel<<<1, 64, 0, stream>>>(g1,b1,m1,v1, g3,b3,m3,v3, g2,b2,m2,v2, sb);
    wpack_kernel<<<(4*576 +255)/256, 256, 0, stream>>>(w1, w1p, 576);
    wpack_kernel<<<(4*1600+255)/256, 256, 0, stream>>>(w3, w3p, 1600);
    wpack_kernel<<<(4*576 +255)/256, 256, 0, stream>>>(w2, w2p, 576);

    offset_conv_kernel<<<nblk, 128, 0, stream>>>(x, w_off, b_off, off1);
    deform_wmma_kernel<<<nblk, 128, 0, stream>>>(x, off1, (const v16h*)w1p, sb, outa);
    conv_wmma_kernel<5,1><<<nblk, 128, 0, stream>>>(x,    (const v16h*)w3p, sb+2*NC, nullptr, outa);
    conv_wmma_kernel<3,2><<<nblk, 128, 0, stream>>>(outa, (const v16h*)w2p, sb+4*NC, x,       out);
}
